// graph_construct_mic_3899830305086
// MI455X (gfx1250) — 8-bit matrix operand order repaired and checked by reading, compiled, not run on hardware
//
#include <hip/hip_runtime.h>
#include <hip/hip_bf16.h>
#include <math.h>

typedef __attribute__((ext_vector_type(2)))  float v2f;
typedef __attribute__((ext_vector_type(8)))  float v8f;
typedef __attribute__((ext_vector_type(8)))  int   v8i;

#define NNODES 512
#define DIM    64
#define NB     16
#define BATCH  32
#define NT     2048
#define LOG64  4.1588830833596715f
#define LOG16  2.772588722239781f
#define THRESH 0.3f

__global__ void k_nodevec(const float* __restrict__ emb,
                          const int*   __restrict__ idx,
                          const float* __restrict__ lin_w,
                          const float* __restrict__ lin_b,
                          float*       __restrict__ nodevec) {
  int n = blockIdx.x;
  int e = threadIdx.x;
  const float* er = emb + (long)idx[n] * DIM;
  const float* wr = lin_w + e * DIM;
  float acc = lin_b[e];
  #pragma unroll 8
  for (int k = 0; k < DIM; ++k) acc = fmaf(er[k], wr[k], acc);
  nodevec[n * DIM + e] = tanhf(acc);
}

__global__ void k_bins(const float* __restrict__ nodevec,
                       unsigned char* __restrict__ bins,
                       float*        __restrict__ S) {
  __shared__ unsigned hist[NB];
  __shared__ float part[NB];
  int n = blockIdx.x, d = threadIdx.x;
  if (d < NB) hist[d] = 0u;
  __syncthreads();
  float v = nodevec[n * DIM + d];
  int b = (int)floorf((v + 1.0f) * 0.5f * (float)NB);
  b = b < 0 ? 0 : (b > NB - 1 ? NB - 1 : b);
  bins[n * DIM + d] = (unsigned char)b;
  atomicAdd(&hist[b], 1u);
  __syncthreads();
  if (d < NB) {
    float rc = (float)hist[d];
    part[d] = (hist[d] > 0u) ? rc * __logf(rc) : 0.0f;
  }
  __syncthreads();
  if (d == 0) {
    float s = 0.0f;
    #pragma unroll
    for (int t = 0; t < NB; ++t) s += part[t];
    S[n] = s;
  }
}

__device__ __forceinline__ unsigned eq4(unsigned w, unsigned tgt4) {
  unsigned y = w ^ tgt4;
  unsigned t = y | (y >> 1);
  t |= t >> 2;
  return ~t & 0x01010101u;
}

__device__ __forceinline__ v8i buildB(const unsigned char* __restrict__ bins,
                                      int j, unsigned tj, int hi) {
  const uint4* bj = (const uint4*)(bins + j * DIM);
  uint4 q0 = bj[hi];
  uint4 q1 = bj[2 + hi];
  v8i b;
  b[0] = (int)eq4(q0.x, tj); b[1] = (int)eq4(q0.y, tj);
  b[2] = (int)eq4(q0.z, tj); b[3] = (int)eq4(q0.w, tj);
  b[4] = (int)eq4(q1.x, tj); b[5] = (int)eq4(q1.y, tj);
  b[6] = (int)eq4(q1.z, tj); b[7] = (int)eq4(q1.w, tj);
  return b;
}

__device__ __forceinline__ float consumeC(v8i c, const float* __restrict__ slut) {
  float acc = 0.0f;
  #pragma unroll
  for (int r = 0; r < 8; ++r) acc += slut[c[r]];
  #pragma unroll
  for (int off = 16; off > 0; off >>= 1)
    acc += __shfl_xor(acc, off, 32);
  return acc;
}

__device__ __forceinline__ void emitAdj(float acc, int lane, float bias,
                                        float scale,
                                        const float* __restrict__ S,
                                        int i, int j,
                                        float* __restrict__ adj) {
  if (lane == 0) {
    float mic = (acc + bias - S[j]) * scale;
    mic = fmaxf(mic, 0.0f);
    adj[i * NNODES + j] = (mic < THRESH) ? 0.0f : mic;
  }
}

__global__ void k_mic(const unsigned char* __restrict__ bins,
                      const float* __restrict__ S,
                      float*       __restrict__ adj) {
  __shared__ unsigned sbins_i[16];
  __shared__ float    slut[DIM + 1];
  const int i    = blockIdx.x;
  const int wave = threadIdx.x >> 5;
  const int lane = threadIdx.x & 31;
  const int lo   = lane & 15;
  const int hi   = lane >> 4;

  if (threadIdx.x < 16)
    sbins_i[threadIdx.x] = ((const unsigned*)(bins + i * DIM))[threadIdx.x];
  if (threadIdx.x <= 64) {
    float t = (float)threadIdx.x;
    slut[threadIdx.x] = (threadIdx.x > 0) ? t * __logf(t) : 0.0f;
  }
  __syncthreads();

  const unsigned trep = (unsigned)lo * 0x01010101u;
  v8i a;
  #pragma unroll
  for (int v = 0; v < 8; ++v) {
    int dw = (v >> 2) * 8 + (v & 3) + 4 * hi;
    a[v] = (int)eq4(sbins_i[dw], trep);
  }

  const float bias  = 64.0f * LOG64 - S[i];
  const float scale = 1.0f / (64.0f * LOG16);
  const int   jbase = wave * 128;
  const v8i   zero  = {0, 0, 0, 0, 0, 0, 0, 0};

  v8i cA = __builtin_amdgcn_wmma_i32_16x16x64_iu8(
      false, a, false, buildB(bins, jbase + 0, trep, hi), zero, false, false);
  v8i cB = __builtin_amdgcn_wmma_i32_16x16x64_iu8(
      false, a, false, buildB(bins, jbase + 1, trep, hi), zero, false, false);

  for (int jj = 0; jj < 126; jj += 2) {
    v8i bA = buildB(bins, jbase + jj + 2, trep, hi);
    emitAdj(consumeC(cA, slut), lane, bias, scale, S, i, jbase + jj, adj);
    cA = __builtin_amdgcn_wmma_i32_16x16x64_iu8(false, a, false, bA,
                                                zero, false, false);
    v8i bB = buildB(bins, jbase + jj + 3, trep, hi);
    emitAdj(consumeC(cB, slut), lane, bias, scale, S, i, jbase + jj + 1, adj);
    cB = __builtin_amdgcn_wmma_i32_16x16x64_iu8(false, a, false, bB,
                                                zero, false, false);
  }
  emitAdj(consumeC(cA, slut), lane, bias, scale, S, i, jbase + 126, adj);
  emitAdj(consumeC(cB, slut), lane, bias, scale, S, i, jbase + 127, adj);
}

__global__ void k_embed(const float* __restrict__ x,
                        const float* __restrict__ nodevec,
                        float*       __restrict__ xout) {
  const int wave  = threadIdx.x >> 5;
  const int lane  = threadIdx.x & 31;
  const int lo    = lane & 15;
  const int hi    = lane >> 4;
  const int etile = (wave & 3) * 16;
  const int ct    = blockIdx.x * 64 + (wave >> 2) * 16 + lo;
  const int batch = blockIdx.y;
  const float* xb = x + (long)batch * NNODES * NT + ct;
  const float* av_base = nodevec + etile + lo;

  v8f acc0 = {0.f, 0.f, 0.f, 0.f, 0.f, 0.f, 0.f, 0.f};
  v8f acc1 = acc0;
  #pragma unroll 2
  for (int kb = 0; kb < NNODES; kb += 8) {
    const int k0 = kb + 2 * hi;
    v2f av, bv;
    av[0] = av_base[(k0 + 0) * DIM];
    av[1] = av_base[(k0 + 1) * DIM];
    bv[0] = xb[(long)(k0 + 0) * NT];
    bv[1] = xb[(long)(k0 + 1) * NT];
    acc0 = __builtin_amdgcn_wmma_f32_16x16x4_f32(false, av, false, bv,
                                                 (short)0, acc0, false, false);
    av[0] = av_base[(k0 + 4) * DIM];
    av[1] = av_base[(k0 + 5) * DIM];
    bv[0] = xb[(long)(k0 + 4) * NT];
    bv[1] = xb[(long)(k0 + 5) * NT];
    acc1 = __builtin_amdgcn_wmma_f32_16x16x4_f32(false, av, false, bv,
                                                 (short)0, acc1, false, false);
  }

  float* ob = xout + (long)batch * DIM * NT + ct;
  #pragma unroll
  for (int r = 0; r < 8; ++r)
    ob[(long)(etile + r + 8 * hi) * NT] = acc0[r] + acc1[r];
}

extern "C" void kernel_launch(void* const* d_in, const int* in_sizes, int n_in,
                              void* d_out, int out_size, void* d_ws, size_t ws_size,
                              hipStream_t stream) {
  const float* x_in  = (const float*)d_in[0];
  const int*   idx   = (const int*)  d_in[1];
  const float* emb_w = (const float*)d_in[2];
  const float* lin_w = (const float*)d_in[3];
  const float* lin_b = (const float*)d_in[4];

  float* adj  = (float*)d_out;
  float* xout = (float*)d_out + NNODES * NNODES;

  char* ws = (char*)d_ws;
  float*         nodevec = (float*)ws;
  unsigned char* bins    = (unsigned char*)(ws + 128 * 1024);
  float*         Sn      = (float*)(ws + 160 * 1024);

  k_nodevec<<<NNODES, DIM, 0, stream>>>(emb_w, idx, lin_w, lin_b, nodevec);
  k_bins   <<<NNODES, DIM, 0, stream>>>(nodevec, bins, Sn);
  k_mic    <<<NNODES, 128, 0, stream>>>(bins, Sn, adj);
  dim3 grid(NT / 64, BATCH);
  k_embed  <<<grid, 512, 0, stream>>>(x_in, nodevec, xout);
}
